// GraphAttentionLayerV2_18949395710593
// MI455X (gfx1250) — matrix operand order repaired and checked by reading, not compiled, not run on hardware
//
#include <hip/hip_runtime.h>
#include <hip/hip_bf16.h>
#include <math.h>

typedef _Float16 f16;
typedef __attribute__((ext_vector_type(16))) _Float16 v16h;
typedef __attribute__((ext_vector_type(8)))  _Float16 v8h;
typedef __attribute__((ext_vector_type(8)))  float    v8f;

#define BN   4
#define NN   512
#define INF_ 128
#define NH   8
#define ND   32
#define OUTF 256   // NH*ND
#define NEG_SLOPE 0.2f

// ---------------------------------------------------------------------------
// Kernel 0: transpose + convert weights to f16:  Wt[out][in] from w[in][out].
// Makes kernel-1 B-fragments contiguous 16-half runs.
// ---------------------------------------------------------------------------
__global__ __launch_bounds__(256)
void gat_prep_weights_kernel(const float* __restrict__ w_l,
                             const float* __restrict__ w_r,
                             f16* __restrict__ wlT,
                             f16* __restrict__ wrT) {
    // 256*128 = 32768 elements per matrix
    for (int idx = blockIdx.x * 256 + threadIdx.x; idx < OUTF * INF_;
         idx += gridDim.x * 256) {
        const int o = idx >> 7;       // out 0..255
        const int k = idx & 127;      // in  0..127
        wlT[o * INF_ + k] = (f16)w_l[k * OUTF + o];
        wrT[o * INF_ + k] = (f16)w_r[k * OUTF + o];
    }
}

// ---------------------------------------------------------------------------
// Kernel 1: g_l = h @ w_l, g_r = h @ w_r   ([2048 x 128] @ [128 x 256])
// One wave per 16x16 output tile; each wave computes BOTH w_l and w_r tiles,
// reusing the A (h) fragments. K loop: 128/32 = 4 x 2 WMMAs per tile.
// B-fragments: contiguous f16 loads from the transposed weight copies.
// ---------------------------------------------------------------------------
__global__ __launch_bounds__(128)
void gat_proj_kernel(const float* __restrict__ h,
                     const f16* __restrict__ wlT,
                     const f16* __restrict__ wrT,
                     float* __restrict__ g_l,
                     float* __restrict__ g_r) {
    const int lane = threadIdx.x & 31;
    const int wv   = (blockIdx.x * (blockDim.x >> 5)) + (threadIdx.x >> 5);
    // tiles: M = 2048/16 = 128, N = 256/16 = 16
    const int mt = wv >> 4;
    const int nt = wv & 15;

    const int row = mt * 16 + (lane & 15);      // A row (M)
    const int kbA = (lane >> 4) * 16;           // A half-lane K base
    const int col = nt * 16 + (lane & 15);      // B col (N = out feature)
    const int kbB = (lane >> 4) * 16;           // B half-lane K base

    v8f cl = {}; v8f cr = {};

    for (int kk = 0; kk < INF_; kk += 32) {
        // ---- A fragment: h[row, kk .. kk+31], f32 -> f16
        v16h a;
        const float* hp = h + row * INF_ + kk + kbA;
        #pragma unroll
        for (int t = 0; t < 8; ++t) {
            a[t]     = (f16)hp[t];
            a[t + 8] = (f16)hp[8 + t];
        }
        // ---- B fragments: contiguous 16 halves Wt[col][kk+kbB .. +15]
        v16h bl, br;
        {
            const f16* wp = wlT + col * INF_ + kk + kbB;
            v8h b0 = *(const v8h*)wp;
            v8h b1 = *(const v8h*)(wp + 8);
            const f16* qp = wrT + col * INF_ + kk + kbB;
            v8h b2 = *(const v8h*)qp;
            v8h b3 = *(const v8h*)(qp + 8);
            #pragma unroll
            for (int t = 0; t < 8; ++t) {
                bl[t] = b0[t]; bl[t + 8] = b1[t];
                br[t] = b2[t]; br[t + 8] = b3[t];
            }
        }
        cl = __builtin_amdgcn_wmma_f32_16x16x32_f16(false, a, false, bl,
                                                    (short)0, cl, false, false);
        cr = __builtin_amdgcn_wmma_f32_16x16x32_f16(false, a, false, br,
                                                    (short)0, cr, false, false);
    }

    // ---- store C/D: vgpr r -> M = r + 8*(lane>=16), N = lane&15
    const int mBase = mt * 16 + ((lane >> 4) << 3);
    const int nCol  = nt * 16 + (lane & 15);
    #pragma unroll
    for (int r = 0; r < 8; ++r) {
        g_l[(mBase + r) * OUTF + nCol] = cl[r];
        g_r[(mBase + r) * OUTF + nCol] = cr[r];
    }
}

// ---------------------------------------------------------------------------
// Kernel 2: pairwise scores + masked row softmax; also emits transposed
// f16 value matrix Vt[bh][d][j] for kernel 3's contiguous B-fragments.
// Block = 256 threads (8 waves) handles one (b, head, tile-of-64-i).
// ---------------------------------------------------------------------------
#define GL_STRIDE 34   // 32 halves + 2 pad -> 17 dwords/row, gcd(17,64)=1

__global__ __launch_bounds__(256)
void gat_score_softmax_kernel(const float* __restrict__ g_l,
                              const float* __restrict__ g_r,
                              const float* __restrict__ w_attn,
                              const int*   __restrict__ adj,
                              f16* __restrict__ attnP,
                              f16* __restrict__ Vt) {
    __shared__ f16   gl_s[NN * GL_STRIDE];   // 34,816 B
    __shared__ float gr_s[64 * ND];          //  8,192 B
    __shared__ float wa_s[ND];

    const int bh    = blockIdx.x >> 3;       // 0..31
    const int itile = blockIdx.x & 7;        // 0..7  (64 i's each)
    const int b     = bh >> 3;
    const int hd    = bh & 7;
    const int tid   = threadIdx.x;

    // cooperative loads
    for (int idx = tid; idx < NN * ND; idx += 256) {
        int j = idx >> 5, d = idx & 31;
        gl_s[j * GL_STRIDE + d] = (f16)g_l[(b * NN + j) * OUTF + hd * ND + d];
    }
    for (int idx = tid; idx < 64 * ND; idx += 256) {
        int il = idx >> 5, d = idx & 31;
        gr_s[idx] = g_r[(b * NN + itile * 64 + il) * OUTF + hd * ND + d];
    }
    // transposed f16 copy of this block's g_r slice: Vt[bh][d][itile*64 + jl]
    // (this block's i-range doubles as the j-range it owns)
    for (int idx = tid; idx < 64 * ND; idx += 256) {
        int d = idx >> 6, jl = idx & 63;     // consecutive tid -> consecutive j
        Vt[(size_t)bh * ND * NN + d * NN + itile * 64 + jl] =
            (f16)g_r[(b * NN + itile * 64 + jl) * OUTF + hd * ND + d];
    }
    if (tid < ND) wa_s[tid] = w_attn[tid];
    __syncthreads();

    const int wave = tid >> 5;
    const int lane = tid & 31;
    f16* rowOutBase = attnP + (size_t)bh * NN * NN;

    for (int ii = 0; ii < 8; ++ii) {
        const int il = wave * 8 + ii;        // local i 0..63
        const int i  = itile * 64 + il;

        float e[16];
        #pragma unroll 1
        for (int jc = 0; jc < 16; ++jc) {
            const int j = jc * 32 + lane;
            const f16* glp = gl_s + j * GL_STRIDE;
            const float* grp = gr_s + il * ND;
            float ev = 0.f;
            #pragma unroll
            for (int d = 0; d < ND; ++d) {
                float x = (float)glp[d] + grp[d];
                x = (x >= 0.f) ? x : NEG_SLOPE * x;
                ev = fmaf(x, wa_s[d], ev);
            }
            const int adjv = adj[i * NN + j];     // adj[0,i,j,0]
            e[jc] = adjv ? ev : -__builtin_inff();
        }

        // softmax over the 512 j's (wave32 shuffle reduction)
        float m = -__builtin_inff();
        #pragma unroll
        for (int jc = 0; jc < 16; ++jc) m = fmaxf(m, e[jc]);
        #pragma unroll
        for (int off = 16; off >= 1; off >>= 1)
            m = fmaxf(m, __shfl_xor(m, off, 32));

        float s = 0.f;
        #pragma unroll
        for (int jc = 0; jc < 16; ++jc) { float p = __expf(e[jc] - m); e[jc] = p; s += p; }
        #pragma unroll
        for (int off = 16; off >= 1; off >>= 1)
            s += __shfl_xor(s, off, 32);

        const float inv = 1.0f / s;
        f16* rp = rowOutBase + (size_t)i * NN;
        #pragma unroll
        for (int jc = 0; jc < 16; ++jc)
            rp[jc * 32 + lane] = (f16)(e[jc] * inv);
    }
}

// ---------------------------------------------------------------------------
// Kernel 3: attn_res[b,i,h,:] = sum_j a[b,i,j,h] * g_r[b,j,h,:]
// Per (b,h): [512 x 512] @ [512 x 32] GEMM on WMMA, K loop 512/32 = 16.
// One wave per 16-row M tile; computes BOTH 16-wide N tiles (d 0..15, 16..31)
// sharing the A fragment. B-fragments are contiguous f16 runs from Vt.
// ---------------------------------------------------------------------------
__global__ __launch_bounds__(128)
void gat_aggregate_kernel(const f16* __restrict__ attnP,
                          const f16* __restrict__ Vt,
                          float* __restrict__ out) {
    const int lane = threadIdx.x & 31;
    const int wv   = (blockIdx.x * (blockDim.x >> 5)) + (threadIdx.x >> 5);
    const int bh   = wv >> 5;                 // 0..31
    const int mt   = wv & 31;                 // 0..31
    const int b    = bh >> 3;
    const int hd   = bh & 7;

    const f16* A  = attnP + (size_t)bh * NN * NN;
    const f16* VB = Vt + (size_t)bh * ND * NN;

    const int row  = mt * 16 + (lane & 15);
    const int kbA  = (lane >> 4) * 16;
    const int colL = lane & 15;               // N tile 0: d = 0..15
    const int kbB  = (lane >> 4) * 16;

    v8f c0 = {}; v8f c1 = {};
    for (int kk = 0; kk < NN; kk += 32) {
        // A fragment: two aligned 16-byte half8 loads
        v16h a;
        const f16* ap = A + (size_t)row * NN + kk + kbA;
        v8h lo = *(const v8h*)(ap);
        v8h hi = *(const v8h*)(ap + 8);
        #pragma unroll
        for (int t = 0; t < 8; ++t) { a[t] = lo[t]; a[t + 8] = hi[t]; }
        __builtin_prefetch(ap + 32, 0, 1);    // next K block (speculative)

        // B fragments: Vt[d][kk+kbB .. +15] contiguous per lane
        v16h b0, b1;
        {
            const f16* vp0 = VB + (size_t)colL * NN + kk + kbB;
            const f16* vp1 = VB + (size_t)(colL + 16) * NN + kk + kbB;
            v8h x0 = *(const v8h*)vp0;
            v8h x1 = *(const v8h*)(vp0 + 8);
            v8h y0 = *(const v8h*)vp1;
            v8h y1 = *(const v8h*)(vp1 + 8);
            #pragma unroll
            for (int t = 0; t < 8; ++t) {
                b0[t] = x0[t]; b0[t + 8] = x1[t];
                b1[t] = y0[t]; b1[t + 8] = y1[t];
            }
        }
        c0 = __builtin_amdgcn_wmma_f32_16x16x32_f16(false, a, false, b0,
                                                    (short)0, c0, false, false);
        c1 = __builtin_amdgcn_wmma_f32_16x16x32_f16(false, a, false, b1,
                                                    (short)0, c1, false, false);
    }

    const int mBase = mt * 16 + ((lane >> 4) << 3);
    const int nCol  = lane & 15;
    #pragma unroll
    for (int r = 0; r < 8; ++r) {
        out[(size_t)(b * NN + mBase + r) * OUTF + hd * ND + nCol]      = c0[r];
        out[(size_t)(b * NN + mBase + r) * OUTF + hd * ND + 16 + nCol] = c1[r];
    }
}

// ---------------------------------------------------------------------------
extern "C" void kernel_launch(void* const* d_in, const int* in_sizes, int n_in,
                              void* d_out, int out_size, void* d_ws, size_t ws_size,
                              hipStream_t stream) {
    const float* h      = (const float*)d_in[0];   // [4,512,128]
    const int*   adj    = (const int*)  d_in[1];   // [1,512,512,1]
    const float* w_l    = (const float*)d_in[2];   // [128,256]
    const float* w_r    = (const float*)d_in[3];   // [128,256]
    const float* w_attn = (const float*)d_in[4];   // [32]
    float* out = (float*)d_out;                    // [4,512,256]

    char* ws = (char*)d_ws;
    size_t off = 0;
    float* g_l   = (float*)(ws + off); off += (size_t)BN * NN * OUTF * 4;       // 2 MB
    float* g_r   = (float*)(ws + off); off += (size_t)BN * NN * OUTF * 4;       // 2 MB
    f16*   attnP = (f16*)  (ws + off); off += (size_t)BN * NH * NN * NN * 2;    // 16 MB
    f16*   Vt    = (f16*)  (ws + off); off += (size_t)BN * NH * ND * NN * 2;    // 1 MB
    f16*   wlT   = (f16*)  (ws + off); off += (size_t)OUTF * INF_ * 2;          // 64 KB
    f16*   wrT   = (f16*)  (ws + off); off += (size_t)OUTF * INF_ * 2;          // 64 KB

    // Kernel 0: weight transpose/convert (tiny)
    gat_prep_weights_kernel<<<32, 256, 0, stream>>>(w_l, w_r, wlT, wrT);

    // Kernel 1: 2048 tiles, 4 waves/block -> 512 blocks
    gat_proj_kernel<<<512, 128, 0, stream>>>(h, wlT, wrT, g_l, g_r);

    // Kernel 2: (b,h) x 8 i-tiles = 256 blocks of 256 threads
    gat_score_softmax_kernel<<<256, 256, 0, stream>>>(g_l, g_r, w_attn, adj, attnP, Vt);

    // Kernel 3: 32 (b,h) x 32 M-tiles = 1024 waves, 4 waves/block -> 256 blocks
    gat_aggregate_kernel<<<256, 128, 0, stream>>>(attnP, Vt, out);
}
